// ST_B_8598524527130
// MI455X (gfx1250) — compile-verified, matrix operand order checked by reading, not run on hardware
//
#include <hip/hip_runtime.h>
#include <hip/hip_bf16.h>

// ---------------------------------------------------------------------------
// MI455X (gfx1250) ST-GCN forward pass.
// Heavy ops = bf16 WMMA GEMMs (v_wmma_f32_16x16x32_bf16, f32 accum).
// Weight tiles loaded via the Tensor Data Mover (tensor_load_to_lds);
// activation tiles gathered branch-free by VALU.
// ---------------------------------------------------------------------------

typedef __attribute__((ext_vector_type(16))) __bf16 v16bf;
typedef __attribute__((ext_vector_type(8)))  float  v8f;
typedef __attribute__((ext_vector_type(4)))  unsigned int v4u;
typedef __attribute__((ext_vector_type(8)))  int    v8i;
typedef __attribute__((ext_vector_type(4)))  int    v4i;

#define RSQ_BN 0.99999500003749978f   // 1/sqrt(1 + 1e-5), eval-mode BN factor

enum { MODE_PLAIN = 0, MODE_TEMPORAL = 1, MODE_GDOWN = 2, MODE_GUP = 3 };
enum { ACT_NONE = 0, ACT_RELU = 1, ACT_LRELU = 2 };

#define TK  32      // K step per LDS tile (matches WMMA K)
#define LDA 40      // padded LDS row stride in halves (80B, 16B aligned)

#if __has_builtin(__builtin_amdgcn_tensor_load_to_lds) && \
    __has_builtin(__builtin_amdgcn_s_wait_tensorcnt)
#define HAVE_TDM 1
#else
#define HAVE_TDM 0
#endif

__device__ inline unsigned short f2bf(float f) {
    union { float f; unsigned int u; } c; c.f = f;
    unsigned int u = c.u;
    u += 0x7FFFu + ((u >> 16) & 1u);   // round-to-nearest-even
    return (unsigned short)(u >> 16);
}

#if HAVE_TDM
// 2D bf16 tile (tile_dim0=32 cols, tile_dim1=64 rows, row stride = K elements)
// -> LDS at byte offset lds_off with 8-half padding per row (stride 40 halves).
__device__ inline void tdm_load_wtile(unsigned lds_off, const unsigned short* gptr,
                                      int K, int Mrem) {
    unsigned long long ga = (unsigned long long)(__SIZE_TYPE__)gptr;
    v4u g0;
    g0[0] = 1u;                                     // count=1 (valid), user mode
    g0[1] = lds_off;                                // lds_addr (bytes)
    g0[2] = (unsigned)(ga & 0xffffffffu);           // global_addr[31:0]
    g0[3] = (unsigned)((ga >> 32) & 0x01ffffffu)    // global_addr[56:32]
            | (2u << 30);                           // type = 2 ("image")
    v8i g1;
    g1[0] = (1 << 16)                               // data_size = 1 (2 bytes)
          | (1 << 20)                               // pad_enable
          | (3 << 22)                               // pad_interval: 16 DWORDs
          | (3 << 25);                              // pad_amount: 4 DWORDs (8 halves)
    g1[1] = (K & 0xffff) << 16;                     // tensor_dim0[15:0]
    g1[2] = ((K >> 16) & 0xffff)                    // tensor_dim0[31:16]
          | ((Mrem & 0xffff) << 16);                // tensor_dim1[15:0]
    g1[3] = ((Mrem >> 16) & 0xffff)                 // tensor_dim1[31:16]
          | (TK << 16);                             // tile_dim0 = 32
    g1[4] = 64;                                     // tile_dim1 = 64, tile_dim2 = 0
    g1[5] = K;                                      // tensor_dim0_stride[31:0]
    g1[6] = 0;
    g1[7] = 0;
    v4i z = {0, 0, 0, 0};
#if __clang_major__ >= 23
    v8i z8 = {0, 0, 0, 0, 0, 0, 0, 0};
    __builtin_amdgcn_tensor_load_to_lds(g0, g1, z, z, z8, 0);
#else
    __builtin_amdgcn_tensor_load_to_lds(g0, g1, z, z, 0);
#endif
}
#endif

// stage one 16-half run of the B tile; CHK only needed for the K tail.
template <int MODE, int VIN, int LG, bool CHK>
__device__ __forceinline__
void stage_B(unsigned short* sB, const float* __restrict__ Xb,
             int k0, int K, int snn, int skb, int st_)
{
    const int TV = 64 * VIN;
    union { unsigned short s[16]; uint4 q[2]; } t;
#pragma unroll
    for (int kk = 0; kk < 16; kk++) {
        int k = k0 + skb + kk;
        float v;
        if (CHK && k >= K) {
            v = 0.f;
        } else if (MODE == MODE_PLAIN || MODE == MODE_GUP) {
            v = Xb[k * TV];
        } else if (MODE == MODE_TEMPORAL) {
            int ci = k / 9, dt = k - ci * 9;
            int tt = st_ + dt - 4;                       // temporal pad 4
            int tc = min(63, max(0, tt));                // clamp -> always-valid addr
            v = Xb[(ci * 64 + tc) * VIN];
            v = (tt == tc) ? v : 0.f;                    // zero the padding
        } else {                                          // MODE_GDOWN
            int c = k / LG, l = k - c * LG;
            v = Xb[c * TV + l];
        }
        t.s[kk] = f2bf(v);
    }
    uint4* d = (uint4*)&sB[snn * LDA + skb];
    d[0] = t.q[0]; d[1] = t.q[1];
}

// Generic tiled GEMM:  D[M,N] = epilogue( sum_k A[M,K]*B[K,N] )
//   A = bf16 W[m*K + k] (pre-converted); B gathered fp32 activations.
// Block: 128 threads (4 waves). 64x64 tile/block, 32x32 tile/wave (2x2 WMMA).
// All launched N are multiples of 64 with exact grids -> no N bound checks.
template <int MODE, int VIN, int LG>
__global__ __launch_bounds__(128)
void k_gemm_wmma(const unsigned short* __restrict__ W, const float* __restrict__ X,
                 float* __restrict__ Y,
                 const float* __restrict__ eb1,     // pre-scale bias, by m
                 const float* __restrict__ escale,  // BN gamma (kernel applies RSQ_BN)
                 const float* __restrict__ eb2,     // BN beta
                 int act, int M, int K, int Cin, int Vout, int inOfs, int outOfs)
{
    __shared__ unsigned short smem[2 * 64 * LDA];  // [0]: A tile, [1]: B tile
    unsigned short* sA = smem;
    unsigned short* sB = smem + 64 * LDA;

    const int tid  = threadIdx.x;
    const int lane = tid & 31;
    const int wave = tid >> 5;
    const int mBlk = blockIdx.y * 64;
    const int nBlk = blockIdx.x * 64;
    const int TV   = 64 * VIN;                     // T * Vin

    // ---- per-thread B staging base (decoded once, 32-bit offsets in-loop) ----
    const int snn = tid >> 1;                      // n row within tile (0..63)
    const int skb = (tid & 1) * 16;                // k sub-run base (0 or 16)
    const int sn  = nBlk + snn;
    const float* Xb;
    int st_ = 0;
    if (MODE == MODE_PLAIN) {
        int sb = sn / TV, sr = sn - sb * TV;
        Xb = X + (size_t)sb * Cin * TV + sr;
    } else if (MODE == MODE_TEMPORAL) {
        int sb = sn / TV, sr = sn - sb * TV;
        st_ = sr / VIN;
        int sv = sr - st_ * VIN;
        Xb = X + (size_t)sb * Cin * TV + sv;
    } else {                                        // GDOWN / GUP
        int sb = sn >> 6, t = sn & 63;
        Xb = X + (size_t)sb * Cin * TV + t * VIN + inOfs;
    }

    v8f acc[2][2];
#pragma unroll
    for (int i = 0; i < 2; i++)
#pragma unroll
        for (int j = 0; j < 2; j++)
            acc[i][j] = (v8f){0.f,0.f,0.f,0.f,0.f,0.f,0.f,0.f};

    const int half = lane >> 4;     // 0: lanes 0-15, 1: lanes 16-31
    const int lr   = lane & 15;
    const int kg   = half * 8;      // k-group base inside 32-wide tile

    for (int k0 = 0; k0 < K; k0 += TK) {
        // ---- A tile: bf16 weight block via Tensor Data Mover ----
#if HAVE_TDM
        if (wave == 0)
            tdm_load_wtile(0 /* sA LDS byte offset */,
                           W + (size_t)mBlk * K + k0, K, M - mBlk);
#else
        {   // fallback: vectorized VALU staging
            const int mm = tid >> 1, m = mBlk + mm;
            union { unsigned short s[16]; uint4 q[2]; } t;
#pragma unroll
            for (int kk = 0; kk < 16; kk++) {
                int k = k0 + skb + kk;
                t.s[kk] = (m < M && k < K) ? W[(size_t)m * K + k] : (unsigned short)0;
            }
            uint4* d = (uint4*)&sA[mm * LDA + skb];
            d[0] = t.q[0]; d[1] = t.q[1];
        }
#endif
        // ---- B tile (n-major [n][k]); uniform fast/tail split ----
        if (k0 + TK <= K)
            stage_B<MODE, VIN, LG, false>(sB, Xb, k0, K, snn, skb, st_);
        else
            stage_B<MODE, VIN, LG, true >(sB, Xb, k0, K, snn, skb, st_);

        // prefetch next activation rows (global_prefetch_b8)
        if ((MODE == MODE_PLAIN || MODE == MODE_GUP) && (k0 + 2 * TK <= K))
            __builtin_prefetch(&Xb[(k0 + TK + skb) * TV], 0, 1);

#if HAVE_TDM
        if (wave == 0) __builtin_amdgcn_s_wait_tensorcnt(0);
#endif
        __syncthreads();

        // ---- fragments: two ds_load_b128 per 16x32/32x16 fragment ----
        union Frag { v16bf v; uint4 q[2]; };
        Frag a[2], b[2];
#pragma unroll
        for (int im = 0; im < 2; im++) {
            const unsigned short* p = &sA[((wave >> 1) * 32 + im * 16 + lr) * LDA + kg];
            a[im].q[0] = *(const uint4*)p;          // K = kg..kg+7
            a[im].q[1] = *(const uint4*)(p + 16);   // K = kg+16..kg+23
        }
#pragma unroll
        for (int in = 0; in < 2; in++) {
            const unsigned short* p = &sB[((wave & 1) * 32 + in * 16 + lr) * LDA + kg];
            b[in].q[0] = *(const uint4*)p;
            b[in].q[1] = *(const uint4*)(p + 16);
        }
#pragma unroll
        for (int im = 0; im < 2; im++)
#pragma unroll
            for (int in = 0; in < 2; in++)
                acc[im][in] = __builtin_amdgcn_wmma_f32_16x16x32_bf16(
                    false, a[im].v, false, b[in].v, (short)0, acc[im][in],
                    false, false);
        __syncthreads();
    }

    // ---- epilogue: bias -> BN -> activation -> scatter ----
    const bool mfull = (mBlk + 64 <= M);
#pragma unroll
    for (int im = 0; im < 2; im++)
#pragma unroll
        for (int in = 0; in < 2; in++) {
            int n = nBlk + (wave & 1) * 32 + in * 16 + lr;
            float* Yb;
            if (MODE == MODE_PLAIN || MODE == MODE_TEMPORAL) {
                int bb = n / TV, r2 = n - bb * TV;
                Yb = Y + (size_t)bb * M * TV + r2;                 // + m*TV
            } else if (MODE == MODE_GDOWN) {
                int bb = n >> 6, t2 = n & 63;
                Yb = Y + ((size_t)bb * M * 64 + t2) * Vout + outOfs;   // + m*64*Vout
            } else {                                                // GUP
                int bb = n >> 6, t2 = n & 63;
                int Och = M / LG;
                Yb = Y + ((size_t)bb * Och * 64 + t2) * Vout + outOfs; // + o*64*Vout + l
            }
            const int mBase = mBlk + (wave >> 1) * 32 + im * 16 + half * 8;
#pragma unroll
            for (int r = 0; r < 8; r++) {
                int m = mBase + r;
                if (!mfull && m >= M) continue;
                float v = acc[im][in][r];
                if (eb1)    v += eb1[m];
                if (escale) v = v * (escale[m] * RSQ_BN) + (eb2 ? eb2[m] : 0.f);
                if (act == ACT_RELU)       v = fmaxf(v, 0.f);
                else if (act == ACT_LRELU) v = v > 0.f ? v : 0.01f * v;

                if (MODE == MODE_PLAIN || MODE == MODE_TEMPORAL) {
                    Yb[m * TV] = v;
                } else if (MODE == MODE_GDOWN) {
                    Yb[m * 64 * Vout] = v;
                } else {                                            // GUP: m = o*LG + l
                    int o = m / LG, l = m - o * LG;
                    Yb[o * 64 * Vout + l] = v;
                }
            }
        }
}

// Y[b,c,t,w] = relu( (sum_v X[b,c,t,v] * A[v,w]) * gamma[c]*RSQ + beta[c] )
__global__ void k_graph_agg(const float* __restrict__ Xg, const float* __restrict__ Adj,
                            float* __restrict__ Y,
                            const float* __restrict__ gma, const float* __restrict__ bta,
                            int C, int V, long total)
{
    long idx = (long)blockIdx.x * blockDim.x + threadIdx.x;
    if (idx >= total) return;
    int w = (int)(idx % V);
    long base = idx - w;                         // start of the v-run
    int c = (int)((base / V / 64) % C);          // T == 64
    const float* row = Xg + base;
    float s = 0.f;
    for (int v = 0; v < V; v++) s += row[v] * Adj[v * V + w];
    s = s * (gma[c] * RSQ_BN) + bta[c];
    Y[idx] = fmaxf(s, 0.f);
}

__global__ void k_add(const float* __restrict__ a, const float* __restrict__ b,
                      float* __restrict__ y, long n)
{
    long i = (long)blockIdx.x * blockDim.x + threadIdx.x;
    if (i < n) y[i] = a[i] + b[i];
}

__global__ void k_cvt_bf16(const float* __restrict__ in, unsigned short* __restrict__ out,
                           long n)
{
    long i = (long)blockIdx.x * blockDim.x + threadIdx.x;
    if (i < n) out[i] = f2bf(in[i]);
}

// ---------------------------------------------------------------------------
extern "C" void kernel_launch(void* const* d_in, const int* in_sizes, int n_in,
                              void* d_out, int out_size, void* d_ws, size_t ws_size,
                              hipStream_t stream)
{
    (void)in_sizes; (void)n_in; (void)out_size; (void)ws_size;
    auto F = [&](int i) { return (const float*)d_in[i]; };

    // Input layout (setup_inputs insertion order):
    const float* x   = F(0);   // [64,3,64,22]
    const float* A22 = F(1);   // [1,22,22]
    const float* A5  = F(2);   // [1,5,5]
    const int ST1 = 3, ST2 = 11, ST3 = 19, ST4 = 27;   // wg,bg,bn1_g,bn1_b,wt,bt,bn2_g,bn2_b
    const int GD1 = 35, GU1 = 45, GD2 = 55, GU2 = 57;  // each entry: w, b

    const int start1[5] = {0, 4, 8, 12, 17};
    const int len1[5]   = {4, 4, 4, 5, 5};

    // workspace carve: fp32 activation buffers, then bf16 weight region
    const size_t F22s = (size_t)64 * 256 * 64 * 22;
    const size_t F5s  = (size_t)64 * 256 * 64 * 5;
    const size_t F1s  = (size_t)64 * 256 * 64;
    const size_t F3s  = (size_t)64 * 3 * 64 * 22;
    float* wsf  = (float*)d_ws;
    float* b22a = wsf;
    float* b22b = b22a + F22s;
    float* b5a  = b22b + F22s;
    float* b5b  = b5a + F5s;
    float* b5c  = b5b + F5s;
    float* b1   = b5c + F5s;
    float* b3a  = b1 + F1s;
    unsigned short* wreg = (unsigned short*)(b3a + F3s);

    // ---- convert all GEMM weights fp32 -> bf16 once ----
    struct WInfo { int idx; size_t n; };
    const WInfo wi[20] = {
        {ST1+0, 768},    {ST1+4, 589824},
        {ST2+0, 65536},  {ST2+4, 589824},
        {ST3+0, 65536},  {ST3+4, 589824},
        {ST4+0, 768},    {ST4+4, 81},
        {GD1+0, 262144}, {GD1+2, 262144}, {GD1+4, 262144}, {GD1+6, 327680}, {GD1+8, 327680},
        {GU1+0, 262144}, {GU1+2, 262144}, {GU1+4, 262144}, {GU1+6, 327680}, {GU1+8, 327680},
        {GD2,   327680}, {GU2,   327680},
    };
    const unsigned short* WB[20];
    {
        size_t ofs = 0;
        for (int i = 0; i < 20; i++) {
            unsigned short* dst = wreg + ofs;
            hipLaunchKernelGGL(k_cvt_bf16, dim3((wi[i].n + 255) / 256), dim3(256), 0,
                               stream, F(wi[i].idx), dst, (long)wi[i].n);
            WB[i] = dst;
            ofs += (wi[i].n + 7) & ~(size_t)7;   // keep 16B alignment
        }
    }
    const unsigned short* wg[4]   = {WB[0], WB[2], WB[4], WB[6]};
    const unsigned short* wt[4]   = {WB[1], WB[3], WB[5], WB[7]};
    const unsigned short* gd1w[5] = {WB[8], WB[9], WB[10], WB[11], WB[12]};
    const unsigned short* gu1w[5] = {WB[13], WB[14], WB[15], WB[16], WB[17]};
    const unsigned short* gd2w = WB[18];
    const unsigned short* gu2w = WB[19];

    const int N22 = 64 * 64 * 22, N5 = 64 * 64 * 5, NT = 64 * 64;

    auto gemm = [&](int mode, int vin, int lg,
                    const unsigned short* Wp, const float* Xp, float* Yp,
                    const float* e1, const float* es, const float* e2, int act,
                    int M, int N, int K, int Cin, int Vout, int inOfs, int outOfs) {
        dim3 grid((N + 63) / 64, (M + 63) / 64);
        dim3 blk(128);
#define LGEMM(MO, VI, LGc) \
        hipLaunchKernelGGL((k_gemm_wmma<MO, VI, LGc>), grid, blk, 0, stream, \
                           Wp, Xp, Yp, e1, es, e2, act, M, K, Cin, Vout, inOfs, outOfs)
        if      (mode == MODE_PLAIN    && vin == 22)            LGEMM(MODE_PLAIN, 22, 1);
        else if (mode == MODE_PLAIN    && vin == 5)             LGEMM(MODE_PLAIN, 5, 1);
        else if (mode == MODE_TEMPORAL && vin == 22)            LGEMM(MODE_TEMPORAL, 22, 1);
        else if (mode == MODE_TEMPORAL && vin == 5)             LGEMM(MODE_TEMPORAL, 5, 1);
        else if (mode == MODE_GDOWN    && vin == 22 && lg == 4) LGEMM(MODE_GDOWN, 22, 4);
        else if (mode == MODE_GDOWN    && vin == 22 && lg == 5) LGEMM(MODE_GDOWN, 22, 5);
        else if (mode == MODE_GDOWN    && vin == 5)             LGEMM(MODE_GDOWN, 5, 5);
        else if (mode == MODE_GUP      && vin == 5)             LGEMM(MODE_GUP, 5, 5);
        else                                                    LGEMM(MODE_GUP, 1, 5);
#undef LGEMM
    };
    auto agg = [&](const float* Xp, const float* Ap, float* Yp,
                   const float* g, const float* b, int C, int V) {
        long total = (long)64 * C * 64 * V;
        hipLaunchKernelGGL(k_graph_agg, dim3((total + 255) / 256), dim3(256), 0, stream,
                           Xp, Ap, Yp, g, b, C, V, total);
    };

    // ---- st1: graph conv (K=3) -> xA + BN + relu -> temporal (K=2304) ----
    gemm(MODE_PLAIN, 22, 0, wg[0], x, b22a, F(ST1+1), nullptr, nullptr, ACT_NONE,
         256, N22, 3, 3, 22, 0, 0);
    agg (b22a, A22, b22b, F(ST1+2), F(ST1+3), 256, 22);
    gemm(MODE_TEMPORAL, 22, 0, wt[0], b22b, b22a, F(ST1+5), F(ST1+6), F(ST1+7), ACT_RELU,
         256, N22, 2304, 256, 22, 0, 0);

    // ---- gdown1: 22 -> 5 nodes (lrelu) ----
    for (int g = 0; g < 5; g++)
        gemm(MODE_GDOWN, 22, len1[g], gd1w[g], b22a, b5a, F(GD1 + 2*g + 1), nullptr,
             nullptr, ACT_LRELU, 256, NT, 256 * len1[g], 256, 5, start1[g], g);

    // ---- st2 on V=5 ----
    gemm(MODE_PLAIN, 5, 0, wg[1], b5a, b5b, F(ST2+1), nullptr, nullptr, ACT_NONE,
         256, N5, 256, 256, 5, 0, 0);
    agg (b5b, A5, b5c, F(ST2+2), F(ST2+3), 256, 5);
    gemm(MODE_TEMPORAL, 5, 0, wt[1], b5c, b5b, F(ST2+5), F(ST2+6), F(ST2+7), ACT_RELU,
         256, N5, 2304, 256, 5, 0, 0);                  // u1 = b5b

    // ---- gdown2: 5 -> 1, gup2: 1 -> 5, residual add ----
    gemm(MODE_GDOWN, 5, 5, gd2w, b5b, b1, F(GD2+1), nullptr, nullptr, ACT_LRELU,
         256, NT, 1280, 256, 1, 0, 0);
    gemm(MODE_GUP, 1, 5, gu2w, b1, b5a, F(GU2+1), nullptr, nullptr, ACT_LRELU,
         1280, NT, 256, 256, 5, 0, 0);
    hipLaunchKernelGGL(k_add, dim3(((long)F5s + 255) / 256), dim3(256), 0, stream,
                       b5a, b5b, b5c, (long)F5s);

    // ---- gup1: 5 -> 22 nodes (lrelu) ----
    for (int g = 0; g < 5; g++)
        gemm(MODE_GUP, 5, len1[g], gu1w[g], b5c, b22b, F(GU1 + 2*g + 1), nullptr,
             nullptr, ACT_LRELU, 256 * len1[g], NT, 256, 256, 22, g, start1[g]);

    // ---- st3 ----
    gemm(MODE_PLAIN, 22, 0, wg[2], b22b, b22a, F(ST3+1), nullptr, nullptr, ACT_NONE,
         256, N22, 256, 256, 22, 0, 0);
    agg (b22a, A22, b22b, F(ST3+2), F(ST3+3), 256, 22);
    gemm(MODE_TEMPORAL, 22, 0, wt[2], b22b, b22a, F(ST3+5), F(ST3+6), F(ST3+7), ACT_RELU,
         256, N22, 2304, 256, 22, 0, 0);

    // ---- st4: H -> 3 ----
    gemm(MODE_PLAIN, 22, 0, wg[3], b22a, b3a, F(ST4+1), nullptr, nullptr, ACT_NONE,
         3, N22, 256, 256, 22, 0, 0);
    agg (b3a, A22, b22b, F(ST4+2), F(ST4+3), 3, 22);
    gemm(MODE_TEMPORAL, 22, 0, wt[3], b22b, (float*)d_out, F(ST4+5), F(ST4+6), F(ST4+7),
         ACT_RELU, 3, N22, 27, 3, 22, 0, 0);
}
